// GraphUNet_36335423324409
// MI455X (gfx1250) — compile-verified, matrix operand order checked by reading, not run on hardware
//
#include <hip/hip_runtime.h>
#include <math.h>

// ---------------------------------------------------------------------------
// GraphUNet forward for MI455X (gfx1250, wave32).
// GEMMs via v_wmma_f32_16x16x32_bf16: inputs pre-converted to bf16, A loaded
// as 2x b128 global loads / chunk, B staged transposed in LDS (2x ds b128).
// Edge aggregation via float atomics (L2-resident working set).
// Top-K via histogram threshold (order-invariant under final mean pool).
// ---------------------------------------------------------------------------

typedef __attribute__((ext_vector_type(16))) __bf16 v16bf;
typedef __attribute__((ext_vector_type(8)))  float  v8f;
typedef __attribute__((ext_vector_type(4)))  unsigned int v4u;
typedef __attribute__((ext_vector_type(8)))  unsigned int v8u;

#define N_NODES 50000
#define N_EDGES 800000
#define C_IN    64
#define C_HID   128
#define C_OUT   64
#define K_KEEP  40000
#define NBINS   4096

static __device__ __forceinline__ unsigned short f2bfu(float f) {
  unsigned u = __builtin_bit_cast(unsigned, f);
  unsigned r = u + 0x7FFFu + ((u >> 16) & 1u);   // round-to-nearest-even
  return (unsigned short)(r >> 16);
}

static __device__ __forceinline__ int score_bin(float s) {
  int b = (int)((s + 1.0f) * (0.5f * (float)NBINS));
  b = b < 0 ? 0 : b;
  return b > (NBINS - 1) ? (NBINS - 1) : b;
}

// fp32 -> bf16 streaming conversion
__global__ void k_cvt_bf16(unsigned short* __restrict__ dst,
                           const float* __restrict__ src, int n) {
  for (int i = blockIdx.x * blockDim.x + threadIdx.x; i < n;
       i += gridDim.x * blockDim.x) dst[i] = f2bfu(src[i]);
}

// --------------------------- WMMA GEMM -------------------------------------
// Y[n,m] = Xb[n,K](bf16) @ W[K,m](f32->bf16 in LDS, transposed).
// n%16==0, m%64==0, K in {64,128} (compile-time -> fully unrolled).
// block = 128 threads (4 waves); wave w -> tile (blockIdx.x*16,
// (blockIdx.y*4+w)*16).
template <int K>
__global__ void k_gemm_bf16(const unsigned short* __restrict__ Xb,
                            const float* __restrict__ W,
                            float* __restrict__ Y, int m) {
  __shared__ unsigned short sWt[128 * 128];   // [m][K] transposed, 32 KB max
  const int tid = threadIdx.x;
  const int total = K * m;
  for (int i = tid; i < total; i += blockDim.x) {
    const int r = i / m, c2 = i - r * m;
    sWt[c2 * K + r] = f2bfu(W[i]);
  }
  __syncthreads();

  const int wave    = tid >> 5;
  const int lane    = tid & 31;
  const int laneLow = lane & 15;
  const int base0   = (lane < 16) ? 0 : 8;   // K sub-offset for this lane half
  const int r0  = blockIdx.x * 16;
  const int c0  = (blockIdx.y * 4 + wave) * 16;
  const int row = r0 + laneLow;
  const int col = c0 + laneLow;

  const unsigned short* arow = Xb + row * K;
  const unsigned short* brow = sWt + col * K;

  v8f c = {};
#pragma unroll
  for (int kk = 0; kk < K; kk += 32) {
    const v4u alo = *(const v4u*)(arow + kk + base0);        // k base0..+7
    const v4u ahi = *(const v4u*)(arow + kk + base0 + 16);   // k base0+16..+23
    const v4u blo = *(const v4u*)(brow + kk + base0);
    const v4u bhi = *(const v4u*)(brow + kk + base0 + 16);
    v8u au, bu;
#pragma unroll
    for (int j = 0; j < 4; ++j) {
      au[j] = alo[j]; au[j + 4] = ahi[j];
      bu[j] = blo[j]; bu[j + 4] = bhi[j];
    }
    const v16bf a = __builtin_bit_cast(v16bf, au);
    const v16bf b = __builtin_bit_cast(v16bf, bu);
    c = __builtin_amdgcn_wmma_f32_16x16x32_bf16(false, a, false, b,
                                                (short)0, c, false, false);
  }
#pragma unroll
  for (int i = 0; i < 8; ++i) {
    const int ro = r0 + i + ((lane < 16) ? 0 : 8);
    Y[ro * m + c0 + laneLow] = c[i];
  }
}

// --------------------------- utility fills ---------------------------------
__global__ void k_fill_f32(float* p, float v, int n) {
  for (int i = blockIdx.x * blockDim.x + threadIdx.x; i < n;
       i += gridDim.x * blockDim.x) p[i] = v;
}
__global__ void k_fill_i32(int* p, int v, int n) {
  for (int i = blockIdx.x * blockDim.x + threadIdx.x; i < n;
       i += gridDim.x * blockDim.x) p[i] = v;
}

// --------------------------- GCN pieces ------------------------------------
__global__ void k_deg_edges(float* deg, const int* __restrict__ dst,
                            const float* __restrict__ mask, int e) {
  for (int i = blockIdx.x * blockDim.x + threadIdx.x; i < e;
       i += gridDim.x * blockDim.x) {
    float w = mask ? mask[i] : 1.0f;
    if (w != 0.0f) atomicAdd(&deg[dst[i]], w);
  }
}

__global__ void k_rsqrt(float* dis, const float* deg, int n) {
  for (int i = blockIdx.x * blockDim.x + threadIdx.x; i < n;
       i += gridDim.x * blockDim.x) dis[i] = rsqrtf(deg[i]);
}

// agg[dst] += dis[src]*dis[dst]*mask * h[src];  grid = E, block = d
__global__ void k_scatter(float* __restrict__ agg, const float* __restrict__ h,
                          const int* __restrict__ src, const int* __restrict__ dst,
                          const float* __restrict__ dis,
                          const float* __restrict__ mask, int d) {
  const int e = blockIdx.x;
  const int f = threadIdx.x;
  const int s = src[e], t = dst[e];
  float w = dis[s] * dis[t];
  if (mask) w *= mask[e];
  if (w != 0.0f) atomicAdd(&agg[t * d + f], w * h[s * d + f]);
}

// agg = [relu](agg + h * dis^2 + bias)
__global__ void k_selfloop(float* __restrict__ agg, const float* __restrict__ h,
                           const float* __restrict__ dis,
                           const float* __restrict__ bias,
                           int n, int d, int relu) {
  const int total = n * d;
  for (int i = blockIdx.x * blockDim.x + threadIdx.x; i < total;
       i += gridDim.x * blockDim.x) {
    const int node = i / d, f = i - node * d;
    const float ds = dis[node];
    float v = agg[i] + h[i] * ds * ds + bias[f];
    agg[i] = relu ? fmaxf(v, 0.0f) : v;
  }
}

// --------------------------- pooling ---------------------------------------
__global__ void k_wnorm(const float* __restrict__ w, float* out) {
  __shared__ float red[128];
  const int t = threadIdx.x;
  const float v = w[t];
  red[t] = v * v;
  __syncthreads();
  for (int s = 64; s > 0; s >>= 1) {
    if (t < s) red[t] += red[t + s];
    __syncthreads();
  }
  if (t == 0) out[0] = rsqrtf(red[0]);
}

// block per node, 128 threads: score = tanh(dot(h[node], pw) * invnorm)
__global__ void k_score(const float* __restrict__ h, const float* __restrict__ pw,
                        const float* __restrict__ invn, float* score) {
  __shared__ float red[128];
  const int node = blockIdx.x, t = threadIdx.x;
  red[t] = h[node * C_HID + t] * pw[t];
  __syncthreads();
  for (int s = 64; s > 0; s >>= 1) {
    if (t < s) red[t] += red[t + s];
    __syncthreads();
  }
  if (t == 0) score[node] = tanhf(red[0] * invn[0]);
}

__global__ void k_hist(const float* __restrict__ score, int* hist, int n) {
  for (int i = blockIdx.x * blockDim.x + threadIdx.x; i < n;
       i += gridDim.x * blockDim.x)
    atomicAdd(&hist[score_bin(score[i])], 1);
}

// single thread: find threshold bin + quota for exactly K_KEEP kept nodes
__global__ void k_thresh(const int* __restrict__ hist, int* ctrs) {
  int cum = 0, bthr = 0, quota = K_KEEP;
  for (int b = NBINS - 1; b >= 0; --b) {
    const int c = hist[b];
    if (cum + c >= K_KEEP) { bthr = b; quota = K_KEEP - cum; break; }
    cum += c;
  }
  ctrs[2] = bthr;
  ctrs[3] = quota;
}

__global__ void k_select(const float* __restrict__ score, int* ctrs,
                         int* __restrict__ perm, float* __restrict__ vals,
                         int* __restrict__ newidx, int n) {
  const int bthr = ctrs[2], quota = ctrs[3];
  for (int i = blockIdx.x * blockDim.x + threadIdx.x; i < n;
       i += gridDim.x * blockDim.x) {
    const float sc = score[i];
    const int b = score_bin(sc);
    int keep = 0;
    if (b > bthr) keep = 1;
    else if (b == bthr) {
      const int q = atomicAdd(&ctrs[1], 1);
      if (q < quota) keep = 1;
    }
    if (keep) {
      const int p = atomicAdd(&ctrs[0], 1);
      perm[p] = i;
      vals[p] = sc;
      newidx[i] = p;
    }
  }
}

// hp[p] = h[perm[p]] * vals[p];  grid = K, block = d
__global__ void k_gather(float* __restrict__ hp, const float* __restrict__ h,
                         const int* __restrict__ perm,
                         const float* __restrict__ vals, int d) {
  const int p = blockIdx.x, f = threadIdx.x;
  hp[p * d + f] = h[perm[p] * d + f] * vals[p];
}

__global__ void k_remap(const int* __restrict__ newidx, const int* __restrict__ src,
                        const int* __restrict__ dst, int* __restrict__ ns,
                        int* __restrict__ nd, float* __restrict__ valid, int e) {
  for (int i = blockIdx.x * blockDim.x + threadIdx.x; i < e;
       i += gridDim.x * blockDim.x) {
    const int s = newidx[src[i]];
    const int t = newidx[dst[i]];
    valid[i] = (s >= 0 && t >= 0) ? 1.0f : 0.0f;
    ns[i] = s < 0 ? 0 : s;
    nd[i] = t < 0 ? 0 : t;
  }
}

// --------------------------- mean pool -------------------------------------
#define MEAN_ROWS 256
__global__ void k_mean(const float* __restrict__ h4, float* out) {
  const int f = threadIdx.x;                 // 64 features
  const int r0 = blockIdx.x * MEAN_ROWS;
  int rend = r0 + MEAN_ROWS;
  rend = rend > K_KEEP ? K_KEEP : rend;
  float s = 0.0f;
  for (int r = r0; r < rend; ++r) s += h4[r * C_OUT + f];
  atomicAdd(&out[f], s * (1.0f / (float)K_KEEP));
}

// ---------------------------------------------------------------------------
extern "C" void kernel_launch(void* const* d_in, const int* in_sizes, int n_in,
                              void* d_out, int out_size, void* d_ws, size_t ws_size,
                              hipStream_t stream) {
  (void)in_sizes; (void)n_in; (void)out_size; (void)ws_size;
  const float* x  = (const float*)d_in[0];
  const int*   ei = (const int*)d_in[1];
  const float* W1 = (const float*)d_in[3];
  const float* b1 = (const float*)d_in[4];
  const float* pw = (const float*)d_in[5];
  const float* W2 = (const float*)d_in[6];
  const float* b2 = (const float*)d_in[7];
  const float* W3 = (const float*)d_in[8];
  const float* b3 = (const float*)d_in[9];
  const float* W4 = (const float*)d_in[10];
  const float* b4 = (const float*)d_in[11];
  const int* src = ei;
  const int* dst = ei + N_EDGES;

  char* p = (char*)d_ws;
  auto carve = [&](size_t bytes) -> char* {
    char* r = p;
    p += (bytes + 255) & ~(size_t)255;
    return r;
  };
  float* hw    = (float*)carve(sizeof(float) * N_NODES * C_HID);  // X@W
  float* agg   = (float*)carve(sizeof(float) * N_NODES * C_HID);  // aggregate/result
  float* hp    = (float*)carve(sizeof(float) * K_KEEP * C_HID);   // pooled features
  unsigned short* xb = (unsigned short*)carve(sizeof(short) * N_NODES * C_HID);
  float* deg   = (float*)carve(sizeof(float) * N_NODES);
  float* dis   = (float*)carve(sizeof(float) * N_NODES);
  float* score = (float*)carve(sizeof(float) * N_NODES);
  float* vals  = (float*)carve(sizeof(float) * K_KEEP);
  float* valid = (float*)carve(sizeof(float) * N_EDGES);
  float* wn    = (float*)carve(sizeof(float) * 16);
  int* newidx  = (int*)carve(sizeof(int) * N_NODES);
  int* perm    = (int*)carve(sizeof(int) * K_KEEP);
  int* ns      = (int*)carve(sizeof(int) * N_EDGES);
  int* nd      = (int*)carve(sizeof(int) * N_EDGES);
  int* hist    = (int*)carve(sizeof(int) * NBINS);
  int* ctrs    = (int*)carve(sizeof(int) * 16);
  float* outf  = (float*)d_out;

  const dim3 GS(2048);   // grid-stride launches
  const dim3 BS(256);

  // ---- conv1: relu(GCN(x, W1, b1)) on original graph -----------------------
  k_fill_f32<<<GS, BS, 0, stream>>>(deg, 1.0f, N_NODES);
  k_deg_edges<<<GS, BS, 0, stream>>>(deg, dst, (const float*)nullptr, N_EDGES);
  k_rsqrt<<<GS, BS, 0, stream>>>(dis, deg, N_NODES);
  k_cvt_bf16<<<GS, BS, 0, stream>>>(xb, x, N_NODES * C_IN);
  k_gemm_bf16<C_IN><<<dim3(N_NODES / 16, C_HID / 64), dim3(128), 0, stream>>>(
      xb, W1, hw, C_HID);
  k_fill_f32<<<GS, BS, 0, stream>>>(agg, 0.0f, N_NODES * C_HID);
  k_scatter<<<dim3(N_EDGES), dim3(C_HID), 0, stream>>>(agg, hw, src, dst, dis,
                                                       (const float*)nullptr, C_HID);
  k_selfloop<<<GS, BS, 0, stream>>>(agg, hw, dis, b1, N_NODES, C_HID, 1);

  // ---- TopKPooling(0.8) ----------------------------------------------------
  k_wnorm<<<dim3(1), dim3(128), 0, stream>>>(pw, wn);
  k_score<<<dim3(N_NODES), dim3(128), 0, stream>>>(agg, pw, wn, score);
  k_fill_i32<<<GS, BS, 0, stream>>>(hist, 0, NBINS);
  k_fill_i32<<<GS, BS, 0, stream>>>(ctrs, 0, 16);
  k_hist<<<GS, BS, 0, stream>>>(score, hist, N_NODES);
  k_thresh<<<dim3(1), dim3(1), 0, stream>>>(hist, ctrs);
  k_fill_i32<<<GS, BS, 0, stream>>>(newidx, -1, N_NODES);
  k_select<<<GS, BS, 0, stream>>>(score, ctrs, perm, vals, newidx, N_NODES);
  k_gather<<<dim3(K_KEEP), dim3(C_HID), 0, stream>>>(hp, agg, perm, vals, C_HID);
  k_remap<<<GS, BS, 0, stream>>>(newidx, src, dst, ns, nd, valid, N_EDGES);

  // pooled-graph degrees (shared by conv2..conv4)
  k_fill_f32<<<GS, BS, 0, stream>>>(deg, 1.0f, K_KEEP);
  k_deg_edges<<<GS, BS, 0, stream>>>(deg, nd, valid, N_EDGES);
  k_rsqrt<<<GS, BS, 0, stream>>>(dis, deg, K_KEEP);

  // ---- conv2: relu(GCN(hp, W2, b2))  [K x 128 -> K x 64] -------------------
  k_cvt_bf16<<<GS, BS, 0, stream>>>(xb, hp, K_KEEP * C_HID);
  k_gemm_bf16<C_HID><<<dim3(K_KEEP / 16, C_OUT / 64), dim3(128), 0, stream>>>(
      xb, W2, hw, C_OUT);
  k_fill_f32<<<GS, BS, 0, stream>>>(agg, 0.0f, K_KEEP * C_OUT);
  k_scatter<<<dim3(N_EDGES), dim3(C_OUT), 0, stream>>>(agg, hw, ns, nd, dis,
                                                       valid, C_OUT);
  k_selfloop<<<GS, BS, 0, stream>>>(agg, hw, dis, b2, K_KEEP, C_OUT, 1);

  // ---- conv3: relu(GCN(h2, W3, b3))  [K x 64 -> K x 128] -------------------
  k_cvt_bf16<<<GS, BS, 0, stream>>>(xb, agg, K_KEEP * C_OUT);
  k_gemm_bf16<C_OUT><<<dim3(K_KEEP / 16, C_HID / 64), dim3(128), 0, stream>>>(
      xb, W3, hw, C_HID);
  k_fill_f32<<<GS, BS, 0, stream>>>(agg, 0.0f, K_KEEP * C_HID);
  k_scatter<<<dim3(N_EDGES), dim3(C_HID), 0, stream>>>(agg, hw, ns, nd, dis,
                                                       valid, C_HID);
  k_selfloop<<<GS, BS, 0, stream>>>(agg, hw, dis, b3, K_KEEP, C_HID, 1);

  // ---- conv4: GCN(h3, W4, b4)  [K x 128 -> K x 64], no relu ----------------
  k_cvt_bf16<<<GS, BS, 0, stream>>>(xb, agg, K_KEEP * C_HID);
  k_gemm_bf16<C_HID><<<dim3(K_KEEP / 16, C_IN / 64), dim3(128), 0, stream>>>(
      xb, W4, hw, C_IN);
  k_fill_f32<<<GS, BS, 0, stream>>>(agg, 0.0f, K_KEEP * C_IN);
  k_scatter<<<dim3(N_EDGES), dim3(C_IN), 0, stream>>>(agg, hw, ns, nd, dis,
                                                      valid, C_IN);
  k_selfloop<<<GS, BS, 0, stream>>>(agg, hw, dis, b4, K_KEEP, C_IN, 0);

  // ---- global mean pool ----------------------------------------------------
  k_fill_f32<<<GS, BS, 0, stream>>>(outf, 0.0f, C_OUT);
  k_mean<<<dim3((K_KEEP + MEAN_ROWS - 1) / MEAN_ROWS), dim3(C_OUT), 0, stream>>>(
      agg, outf);
}
